// NEDModule_12635793784905
// MI455X (gfx1250) — compile-verified, matrix operand order checked by reading, not run on hardware
//
#include <hip/hip_runtime.h>
#include <hip/hip_bf16.h>
#include <stdint.h>

// Problem dims (from reference)
#define B_      32
#define S_      128
#define IN_DIM_ 1024
#define H_      1024
#define NH_     8
#define DH_     128
#define OUT_D_  256
#define K_      10
#define L_      2
#define DFF_    2048
#define TOK_    (B_ * (S_ + 1))              // 4128 token rows
#define PRED_SZ ((size_t)B_ * S_ * OUT_D_)   // 1048576
#define CROWS_  (B_ * S_ * K_)               // 40960 candidate rows

typedef __attribute__((ext_vector_type(16))) __bf16 v16bf;
typedef __attribute__((ext_vector_type(8)))  __bf16 v8bf;
typedef __attribute__((ext_vector_type(4)))  __bf16 v4bf;
typedef __attribute__((ext_vector_type(8)))  float  v8f;

// Hardware f32 -> bf16 conversion (lowers to v_cvt_*bf16_f32 on gfx1250)
__device__ __forceinline__ __bf16 f2bf(float f) { return (__bf16)f; }

// ---------------------------------------------------------------------------
// Generic GEMM: C = act(A(MxK) * W(KxN) + bias [+ resid]), bf16 WMMA, f32 acc.
// Block: 256 threads (8 waves). Block tile 128(M) x 128(N). Wave tile 32x64.
// Preconditions used (true for every call below):
//   Kdim % 32 == 0, N % 128 == 0, grid.x == N/128 exact. Only M needs guards.
// ---------------------------------------------------------------------------
template<int ACT, bool RES>
__global__ __launch_bounds__(256) void gemm_wmma_kernel(
    const float* __restrict__ A, const float* __restrict__ W,
    const float* __restrict__ bias, const float* __restrict__ resid,
    float* __restrict__ C, int M, int N, int Kdim)
{
  __shared__ alignas(16) __bf16 As[128 * 40];   // [row][k], stride 40 elems (80B)
  __shared__ alignas(16) __bf16 Bs[128 * 40];   // transposed: [n][k]

  const int tid  = threadIdx.x;
  const int lane = tid & 31;
  const int wave = tid >> 5;
  const int wm   = wave >> 1;            // 0..3  (M direction, 32 rows each)
  const int wn   = wave & 1;             // 0..1  (N direction, 64 cols each)
  const int m0   = blockIdx.y * 128;
  const int n0   = blockIdx.x * 128;
  const int r    = lane & 15;            // row/col within a 16-tile
  const int kh   = (lane >> 4) << 3;     // K-half offset: 0 or 8

  // staging coordinates (fixed per thread)
  const int a_kg = (tid & 7) * 4;        // k-group of 4 within 32
  const int a_rw = tid >> 3;             // base row (0..31), +32 per iter
  const int b_n  = tid & 127;            // column within 128
  const int b_kg0= (tid >> 7) * 4;       // 0 or 4, +8 per iter

  v8f acc[2][4] = {};

  for (int k0 = 0; k0 < Kdim; k0 += 32) {
    // ---- issue ALL staging global loads first (max memory-level parallelism)
    float4 av[4];
    #pragma unroll
    for (int ii = 0; ii < 4; ++ii) {
      int gr = m0 + a_rw + ii * 32; if (gr > M - 1) gr = M - 1;  // clamp: OOB rows unused
      av[ii] = *(const float4*)&A[(size_t)gr * Kdim + (k0 + a_kg)];
    }
    float bv[4][4];
    #pragma unroll
    for (int ii = 0; ii < 4; ++ii) {
      int kg = b_kg0 + ii * 8;
      const float* wp = &W[(size_t)(k0 + kg) * N + (n0 + b_n)];
      bv[ii][0] = wp[0];
      bv[ii][1] = wp[(size_t)N];
      bv[ii][2] = wp[(size_t)2 * N];
      bv[ii][3] = wp[(size_t)3 * N];
    }
    // ---- convert + store to LDS
    #pragma unroll
    for (int ii = 0; ii < 4; ++ii) {
      v4bf t; t[0] = f2bf(av[ii].x); t[1] = f2bf(av[ii].y);
              t[2] = f2bf(av[ii].z); t[3] = f2bf(av[ii].w);
      *(v4bf*)&As[(a_rw + ii * 32) * 40 + a_kg] = t;
    }
    #pragma unroll
    for (int ii = 0; ii < 4; ++ii) {
      v4bf t; t[0] = f2bf(bv[ii][0]); t[1] = f2bf(bv[ii][1]);
              t[2] = f2bf(bv[ii][2]); t[3] = f2bf(bv[ii][3]);
      *(v4bf*)&Bs[b_n * 40 + (b_kg0 + ii * 8)] = t;
    }
    __syncthreads();

    // ---- prefetch next K tile (uniform branch, no divergence)
    if (k0 + 32 < Kdim) {
      int gr = m0 + a_rw; if (gr > M - 1) gr = M - 1;
      __builtin_prefetch(&A[(size_t)gr * Kdim + (k0 + 32 + a_kg)], 0, 1);
      __builtin_prefetch(&W[(size_t)(k0 + 32 + b_kg0) * N + (n0 + b_n)], 0, 1);
    }

    // ---- fragments per ISA 16-bit layout:
    // lane-half kh holds K = kh..kh+7 (VGPR0-3) and 16+kh..+7 (VGPR4-7)
    v16bf afrag[2], bfrag[4];
    #pragma unroll
    for (int mi = 0; mi < 2; ++mi) {
      int row = wm * 32 + mi * 16 + r;
      v8bf lo = *(const v8bf*)&As[row * 40 + kh];
      v8bf hi = *(const v8bf*)&As[row * 40 + 16 + kh];
      #pragma unroll
      for (int j = 0; j < 8; ++j) { afrag[mi][j] = lo[j]; afrag[mi][j + 8] = hi[j]; }
    }
    #pragma unroll
    for (int ni = 0; ni < 4; ++ni) {
      int col = wn * 64 + ni * 16 + r;
      v8bf lo = *(const v8bf*)&Bs[col * 40 + kh];
      v8bf hi = *(const v8bf*)&Bs[col * 40 + 16 + kh];
      #pragma unroll
      for (int j = 0; j < 8; ++j) { bfrag[ni][j] = lo[j]; bfrag[ni][j + 8] = hi[j]; }
    }

    #pragma unroll
    for (int mi = 0; mi < 2; ++mi)
      #pragma unroll
      for (int ni = 0; ni < 4; ++ni)
        acc[mi][ni] = __builtin_amdgcn_wmma_f32_16x16x32_bf16(
            false, afrag[mi], false, bfrag[ni], (short)0, acc[mi][ni], false, false);

    __syncthreads();
  }

  // ---- epilogue: C/D layout — VGPR v holds row (lane>>4)*8+v, col lane&15
  const int rowHalf = (lane >> 4) << 3;
  const int colIn   = lane & 15;
  #pragma unroll
  for (int mi = 0; mi < 2; ++mi) {
    #pragma unroll
    for (int ni = 0; ni < 4; ++ni) {
      int colg = n0 + wn * 64 + ni * 16 + colIn;      // always < N
      float bval = bias[colg];
      #pragma unroll
      for (int v = 0; v < 8; ++v) {
        int rowg = m0 + wm * 32 + mi * 16 + rowHalf + v;
        if (rowg >= M) continue;
        float x = acc[mi][ni][v] + bval;
        if (RES) x += resid[(size_t)rowg * N + colg];
        if (ACT == 1) x = fmaxf(x, 0.0f);
        C[(size_t)rowg * N + colg] = x;
      }
    }
  }
}

// ---------------------------------------------------------------------------
// concat([ctx, x], axis=1) -> h (TOK_, IN_DIM_)
// ---------------------------------------------------------------------------
__global__ __launch_bounds__(256) void concat_kernel(
    const float* __restrict__ x, const float* __restrict__ ctx, float* __restrict__ h)
{
  size_t i = (size_t)blockIdx.x * 256 + threadIdx.x;
  if (i >= (size_t)TOK_ * IN_DIM_) return;
  int c = (int)(i & (IN_DIM_ - 1));
  size_t row = i >> 10;
  int b = (int)(row / (S_ + 1)), t = (int)(row % (S_ + 1));
  h[i] = (t == 0) ? ctx[(size_t)b * IN_DIM_ + c]
                  : x[((size_t)b * S_ + (t - 1)) * IN_DIM_ + c];
}

// ---------------------------------------------------------------------------
// Attention exactly as the reference einsums: s-axis = 32, b-axis = 129.
// One block per (head, b); softmax over 32.
// ---------------------------------------------------------------------------
__global__ __launch_bounds__(256) void attn_kernel(
    const float* __restrict__ qkv, float* __restrict__ O)
{
  __shared__ float qs[32 * 128], ks[32 * 128], vs[32 * 128];
  __shared__ float sc[32 * 33];
  const int h  = blockIdx.x;     // 0..7
  const int bt = blockIdx.y;     // 0..128
  const int tid = threadIdx.x;

  for (int i = tid; i < 32 * 128; i += 256) {
    int s = i >> 7, d = i & 127;
    size_t base = ((size_t)(s * (S_ + 1) + bt)) * 3 * H_ + h * DH_ + d;
    qs[i] = qkv[base];
    ks[i] = qkv[base + H_];
    vs[i] = qkv[base + 2 * H_];
  }
  __syncthreads();

  const float scale = 0.08838834764831845f;  // 1/sqrt(128)
  for (int p = tid; p < 32 * 32; p += 256) {
    int s = p >> 5, t = p & 31;
    float acc = 0.0f;
    for (int d = 0; d < 128; ++d) acc += qs[s * 128 + d] * ks[t * 128 + d];
    sc[s * 33 + t] = acc * scale;
  }
  __syncthreads();

  if (tid < 32) {
    float mx = -3.4e38f;
    for (int t = 0; t < 32; ++t) mx = fmaxf(mx, sc[tid * 33 + t]);
    float sum = 0.0f;
    for (int t = 0; t < 32; ++t) { float e = __expf(sc[tid * 33 + t] - mx); sc[tid * 33 + t] = e; sum += e; }
    float inv = 1.0f / sum;
    for (int t = 0; t < 32; ++t) sc[tid * 33 + t] *= inv;
  }
  __syncthreads();

  for (int i = tid; i < 32 * 128; i += 256) {
    int s = i >> 7, d = i & 127;
    float acc = 0.0f;
    for (int t = 0; t < 32; ++t) acc += sc[s * 33 + t] * vs[t * 128 + d];
    O[((size_t)(s * (S_ + 1) + bt)) * H_ + h * DH_ + d] = acc;
  }
}

// ---------------------------------------------------------------------------
// LayerNorm over last dim (N=1024), one block per row.
// ---------------------------------------------------------------------------
__global__ __launch_bounds__(256) void ln_kernel(
    const float* __restrict__ X, const float* __restrict__ g,
    const float* __restrict__ bshift, float* __restrict__ Y, int N)
{
  __shared__ float red[256];
  const int tid = threadIdx.x;
  const size_t base = (size_t)blockIdx.x * N;

  float s = 0.0f;
  for (int c = tid; c < N; c += 256) s += X[base + c];
  red[tid] = s; __syncthreads();
  for (int off = 128; off > 0; off >>= 1) { if (tid < off) red[tid] += red[tid + off]; __syncthreads(); }
  float mean = red[0] / N;
  __syncthreads();

  float v = 0.0f;
  for (int c = tid; c < N; c += 256) { float d = X[base + c] - mean; v += d * d; }
  red[tid] = v; __syncthreads();
  for (int off = 128; off > 0; off >>= 1) { if (tid < off) red[tid] += red[tid + off]; __syncthreads(); }
  float rstd = rsqrtf(red[0] / N + 1e-5f);

  for (int c = tid; c < N; c += 256)
    Y[base + c] = (X[base + c] - mean) * rstd * g[c] + bshift[c];
}

// pred = out[:, 1:, :]
__global__ __launch_bounds__(256) void copy_pred_kernel(
    const float* __restrict__ o, float* __restrict__ dst)
{
  size_t i = (size_t)blockIdx.x * 256 + threadIdx.x;
  int c = (int)(i & (OUT_D_ - 1));
  size_t row = i >> 8;                 // b*128 + s
  int s = (int)(row & 127);
  int b = (int)(row >> 7);
  dst[i] = o[((size_t)(b * (S_ + 1) + s + 1)) * OUT_D_ + c];
}

// gather KB, build D = cand - pred, and write cand into cand_out[..., 1:257]
__global__ __launch_bounds__(256) void build_d_kernel(
    const int* __restrict__ idxs, const float* __restrict__ KB,
    const float* __restrict__ o, float* __restrict__ Dbuf, float* __restrict__ dout)
{
  int rr = blockIdx.x;                 // b*1280 + s*10 + k
  int c  = threadIdx.x;
  int idx = idxs[rr]; if (idx < 0) idx = 0;
  int s = (rr / K_) % S_;
  int b = rr / (S_ * K_);
  float cand = KB[(size_t)idx * OUT_D_ + c];
  float pred = o[((size_t)(b * (S_ + 1) + s + 1)) * OUT_D_ + c];
  Dbuf[(size_t)rr * OUT_D_ + c] = cand - pred;
  dout[PRED_SZ + (size_t)rr * (OUT_D_ + 1) + 1 + c] = cand;
}

// ds = relu(D@W1+b1) @ W2 + b2 (second GEMM is a length-256 dot per row)
__global__ __launch_bounds__(256) void ds_final_kernel(
    const float* __restrict__ Hb, const float* __restrict__ W2v,
    const float* __restrict__ b2v, float* __restrict__ dout)
{
  int rr = blockIdx.x * 256 + threadIdx.x;
  if (rr >= CROWS_) return;
  float acc = b2v[0];
  for (int c = 0; c < OUT_D_; ++c) acc += Hb[(size_t)rr * OUT_D_ + c] * W2v[c];
  dout[PRED_SZ + (size_t)rr * (OUT_D_ + 1)] = acc;
}

// ---------------------------------------------------------------------------
static inline void gemm(const float* A, const float* W, const float* bias,
                        const float* resid, float* C, int M, int N, int K,
                        int act, hipStream_t s)
{
  dim3 grid(N / 128, (M + 127) / 128), blk(256);
  if (resid) {
    if (act) gemm_wmma_kernel<1, true ><<<grid, blk, 0, s>>>(A, W, bias, resid, C, M, N, K);
    else     gemm_wmma_kernel<0, true ><<<grid, blk, 0, s>>>(A, W, bias, resid, C, M, N, K);
  } else {
    if (act) gemm_wmma_kernel<1, false><<<grid, blk, 0, s>>>(A, W, bias, resid, C, M, N, K);
    else     gemm_wmma_kernel<0, false><<<grid, blk, 0, s>>>(A, W, bias, resid, C, M, N, K);
  }
}

extern "C" void kernel_launch(void* const* d_in, const int* in_sizes, int n_in,
                              void* d_out, int out_size, void* d_ws, size_t ws_size,
                              hipStream_t stream)
{
  const float* x       = (const float*)d_in[0];
  const float* ctx     = (const float*)d_in[1];
  const int*   indices = (const int*)  d_in[2];
  const float* KB      = (const float*)d_in[3];
  const float* W_in    = (const float*)d_in[4];
  const float* b_in    = (const float*)d_in[5];
  const float* Wqkv    = (const float*)d_in[6];
  const float* bqkv    = (const float*)d_in[7];
  const float* Wo      = (const float*)d_in[8];
  const float* bo      = (const float*)d_in[9];
  const float* ln1_g   = (const float*)d_in[10];
  const float* ln1_b   = (const float*)d_in[11];
  const float* W1      = (const float*)d_in[12];
  const float* b1      = (const float*)d_in[13];
  const float* W2      = (const float*)d_in[14];
  const float* b2      = (const float*)d_in[15];
  const float* ln2_g   = (const float*)d_in[16];
  const float* ln2_b   = (const float*)d_in[17];
  const float* W_out   = (const float*)d_in[18];
  const float* b_out   = (const float*)d_in[19];
  const float* ds_W1   = (const float*)d_in[20];
  const float* ds_b1   = (const float*)d_in[21];
  const float* ds_W2   = (const float*)d_in[22];
  const float* ds_b2   = (const float*)d_in[23];
  (void)in_sizes; (void)n_in; (void)out_size; (void)ws_size;

  float* out = (float*)d_out;
  float* ws  = (float*)d_ws;

  const size_t SZ_H = (size_t)TOK_ * H_;        // 4,227,072 floats
  float* H0   = ws;
  float* H1   = ws + SZ_H;
  float* H2   = ws + 2 * SZ_H;
  float* QKV  = ws + 3 * SZ_H;                  // TOK_*3H
  float* FF   = QKV + (size_t)TOK_ * 3 * H_;    // TOK_*DFF (also attn O)
  float* OUTB = FF + (size_t)TOK_ * DFF_;       // TOK_*OUT_D
  float* DBUF = ws;                             // reuse (H* dead by then)
  float* HBUF = QKV;                            // reuse (QKV dead by then)

  // h_in = concat(ctx, x)
  concat_kernel<<<((size_t)TOK_ * IN_DIM_ + 255) / 256, 256, 0, stream>>>(x, ctx, H0);
  // h = relu(h_in @ W_in + b_in)
  gemm(H0, W_in, b_in, nullptr, H1, TOK_, H_, IN_DIM_, 1, stream);

  for (int l = 0; l < L_; ++l) {
    gemm(H1, Wqkv + (size_t)l * H_ * 3 * H_, bqkv + (size_t)l * 3 * H_,
         nullptr, QKV, TOK_, 3 * H_, H_, 0, stream);
    attn_kernel<<<dim3(NH_, S_ + 1), 256, 0, stream>>>(QKV, FF);
    // h + O@Wo + bo
    gemm(FF, Wo + (size_t)l * H_ * H_, bo + (size_t)l * H_, H1, H2, TOK_, H_, H_, 0, stream);
    ln_kernel<<<TOK_, 256, 0, stream>>>(H2, ln1_g + (size_t)l * H_, ln1_b + (size_t)l * H_, H0, H_);
    // ff = relu(h@W1+b1)
    gemm(H0, W1 + (size_t)l * H_ * DFF_, b1 + (size_t)l * DFF_, nullptr, FF, TOK_, DFF_, H_, 1, stream);
    // h + ff@W2 + b2
    gemm(FF, W2 + (size_t)l * DFF_ * H_, b2 + (size_t)l * H_, H0, H2, TOK_, H_, DFF_, 0, stream);
    ln_kernel<<<TOK_, 256, 0, stream>>>(H2, ln2_g + (size_t)l * H_, ln2_b + (size_t)l * H_, H1, H_);
  }

  // out = h @ W_out + b_out
  gemm(H1, W_out, b_out, nullptr, OUTB, TOK_, OUT_D_, H_, 0, stream);

  copy_pred_kernel<<<(unsigned)(PRED_SZ / 256), 256, 0, stream>>>(OUTB, out);
  build_d_kernel<<<CROWS_, 256, 0, stream>>>(indices, KB, OUTB, DBUF, out);
  // hidden = relu(D @ ds_W1 + ds_b1)  (40960 x 256 x 256)
  gemm(DBUF, ds_W1, ds_b1, nullptr, HBUF, CROWS_, OUT_D_, OUT_D_, 1, stream);
  ds_final_kernel<<<(CROWS_ + 255) / 256, 256, 0, stream>>>(HBUF, ds_W2, ds_b2, out);
}
